// HFP_32083405701516
// MI455X (gfx1250) — matrix operand order repaired and checked by reading, not compiled, not run on hardware
//
#include <hip/hip_runtime.h>
#include <hip/hip_bf16.h>
#include <math.h>

// ---------------------------------------------------------------------------
// Problem constants (match reference)
// ---------------------------------------------------------------------------
#define BB 4
#define CC 128
#define HH_ 256
#define WW 256
#define BC (BB * CC)          // 512 planes
#define NBLK ((HH_/2) * (WW/2)) // 16384 2x2 blocks per plane
#define PLANE (HH_ * WW)      // 65536 elements per plane
#define HIDDEN 16

typedef __attribute__((ext_vector_type(16))) _Float16 v16h;
typedef __attribute__((ext_vector_type(8)))  float    v8f;

__device__ __forceinline__ float sigmoidf_(float v) { return 1.f / (1.f + expf(-v)); }

// ---------------------------------------------------------------------------
// Kernel A: per-(b,c) plane reductions.
//   msum[k*BC+p] = sum over plane of |band_k|  (k = 0:LH, 1:HL, 2:HH)
//   gsum[p]      = sum over plane of x         (== HW * mean(X_re), see analysis)
// One block per plane, 256 threads, coalesced float2 row-pair loads.
// ---------------------------------------------------------------------------
__global__ void hfp_kA_plane_stats(const float* __restrict__ x,
                                   float* __restrict__ msum,
                                   float* __restrict__ gsum) {
    const int p = blockIdx.x;
    const float* xp = x + (size_t)p * PLANE;
    float sLH = 0.f, sHL = 0.f, sHH = 0.f, sX = 0.f;
    for (int i = threadIdx.x; i < NBLK; i += 256) {
        const int h2 = i >> 7, w2 = i & 127;
        const float* r0 = xp + (2 * h2) * WW + 2 * w2;
        const float2 t0 = *(const float2*)r0;
        const float2 t1 = *(const float2*)(r0 + WW);
        const float a = t0.x, b = t0.y, c = t1.x, d = t1.y;
        sLH += fabsf(0.5f * (-a + b - c + d));
        sHL += fabsf(0.5f * (-a - b + c + d));
        sHH += fabsf(0.5f * ( a - b - c + d));
        sX  += a + b + c + d;
    }
    __shared__ float red[4][256];
    red[0][threadIdx.x] = sLH; red[1][threadIdx.x] = sHL;
    red[2][threadIdx.x] = sHH; red[3][threadIdx.x] = sX;
    __syncthreads();
    for (int s = 128; s > 0; s >>= 1) {
        if (threadIdx.x < s) {
            #pragma unroll
            for (int k = 0; k < 4; k++)
                red[k][threadIdx.x] += red[k][threadIdx.x + s];
        }
        __syncthreads();
    }
    if (threadIdx.x == 0) {
        msum[0 * BC + p] = red[0][0];
        msum[1 * BC + p] = red[1][0];
        msum[2 * BC + p] = red[2][0];
        gsum[p] = red[3][0];
    }
}

// ---------------------------------------------------------------------------
// Kernel B: per-(b,h2,w2) band attention.
//   s_k = mean_c | soft_threshold(band_k, thr_k[c]) * g_k[c] |
//       = mean_c relu(|band_k| - thr_k[c]) * g_k[c]     (g_k > 0)
//   attn_k = softmax over k, stored as 3 planes of (B,128,128).
// x is L2-resident after kernel A, so the channel-loop re-read is cheap.
// ---------------------------------------------------------------------------
__global__ void hfp_kB_attn(const float* __restrict__ x,
                            const float* __restrict__ theta,
                            const float* __restrict__ dgate,
                            const float* __restrict__ msum,
                            float* __restrict__ attn) {
    __shared__ float thr[3][CC];
    __shared__ float gg[3][CC];
    const int b = blockIdx.x >> 6;
    if (threadIdx.x < CC) {
        const int c = threadIdx.x;
        #pragma unroll
        for (int k = 0; k < 3; k++) {
            const float t = sigmoidf_(theta[k * CC + c]);
            const float m = msum[k * BC + b * CC + c] * (1.f / (float)NBLK) + 1e-6f;
            thr[k][c] = t * m;
            gg[k][c]  = sigmoidf_(dgate[k * CC + c]);
        }
    }
    __syncthreads();
    const int idx = (blockIdx.x & 63) * 256 + threadIdx.x;   // 0..16383
    const int h2 = idx >> 7, w2 = idx & 127;
    const float* xb = x + (size_t)b * CC * PLANE + (2 * h2) * WW + 2 * w2;
    float s0 = 0.f, s1 = 0.f, s2 = 0.f;
    for (int c = 0; c < CC; c++) {
        const float* r0 = xb + (size_t)c * PLANE;
        const float2 t0 = *(const float2*)r0;
        const float2 t1 = *(const float2*)(r0 + WW);
        const float a = t0.x, bb_ = t0.y, cc_ = t1.x, dd = t1.y;
        const float LH = 0.5f * (-a + bb_ - cc_ + dd);
        const float HL = 0.5f * (-a - bb_ + cc_ + dd);
        const float HHv = 0.5f * ( a - bb_ - cc_ + dd);
        s0 += fmaxf(fabsf(LH)  - thr[0][c], 0.f) * gg[0][c];
        s1 += fmaxf(fabsf(HL)  - thr[1][c], 0.f) * gg[1][c];
        s2 += fmaxf(fabsf(HHv) - thr[2][c], 0.f) * gg[2][c];
    }
    s0 *= (1.f / (float)CC); s1 *= (1.f / (float)CC); s2 *= (1.f / (float)CC);
    const float mx = fmaxf(s0, fmaxf(s1, s2));
    const float e0 = expf(s0 - mx), e1 = expf(s1 - mx), e2 = expf(s2 - mx);
    const float inv = 1.f / (e0 + e1 + e2);
    attn[(size_t)(0 * BB + b) * NBLK + idx] = e0 * inv;
    attn[(size_t)(1 * BB + b) * NBLK + idx] = e1 * inv;
    attn[(size_t)(2 * BB + b) * NBLK + idx] = e2 * inv;
}

// ---------------------------------------------------------------------------
// Kernel D: the FC + SE head on the WMMA path.
//   gap[b][c]  = gsum[p] / 65536   (== mean(X_re), proved in analysis)
//   logits     = gap @ fc_w^T + fc_b   -> softmax -> wsoft  (GEMM1, 8 waves)
//   z          = relu(gap @ se_w1^T + se_b1)                (GEMM2, wave 0)
//   w_se       = sigmoid(z @ se_w2^T + se_b2)               (GEMM3, 8 waves)
// Batch (4) padded to M=16; all GEMMs run as v_wmma_f32_16x16x32_f16 with
// f32 accumulation. Operand packing follows ISA 7.12.2:
//   A 16x32 f16 : lane L holds M=L%16, Kbase=(L>=16)*8;
//                 element e -> K = Kbase + (e&7) + (e>>3)*16
//   B 32x16 f16 : lane L holds N=L%16; element e -> K = (L>=16)*8 + (e&7) + (e>>3)*16
//   C/D 16x16 f32: VGPR r, lane L -> M = r + (L>=16)*8, N = L%16
// ---------------------------------------------------------------------------
__global__ void hfp_kD_head(const float* __restrict__ gsum,
                            const float* __restrict__ fc_w,
                            const float* __restrict__ fc_b,
                            const float* __restrict__ se_w1,
                            const float* __restrict__ se_b1,
                            const float* __restrict__ se_w2,
                            const float* __restrict__ se_b2,
                            float* __restrict__ wsoft,
                            float* __restrict__ wse) {
    __shared__ _Float16 Asm[16][CC];       // gap, padded to 16 rows
    __shared__ _Float16 Zsm[16][32];       // z, padded to 16x32 (K pad)
    __shared__ float    logits[BB][CC];
    const int tid  = threadIdx.x;
    const int wave = tid >> 5;
    const int lane = tid & 31;

    for (int i = tid; i < 16 * CC; i += 256) {
        const int m = i >> 7, c = i & 127;
        const float v = (m < BB) ? gsum[m * CC + c] * (1.f / (float)PLANE) : 0.f;
        Asm[m][c] = (_Float16)v;
    }
    for (int i = tid; i < 16 * 32; i += 256) Zsm[i >> 5][i & 31] = (_Float16)0.f;
    __syncthreads();

    const int M   = lane & 15;
    const int kbA = (lane >> 4) * 8;
    const int N   = lane & 15;
    const int kbB = (lane >> 4) * 8;
    const int j0  = wave * 16;

    // ---- GEMM1: logits = gap @ fc_w^T (each wave owns 16 output channels)
    v8f acc = {};
    for (int kk = 0; kk < CC; kk += 32) {
        v16h av, bv;
        #pragma unroll
        for (int e = 0; e < 16; e++) {
            av[e] = Asm[M][kk + kbA + (e & 7) + ((e >> 3) << 4)];
            bv[e] = (_Float16)fc_w[(j0 + N) * CC + kk + kbB + (e & 7) + ((e >> 3) << 4)];
        }
        acc = __builtin_amdgcn_wmma_f32_16x16x32_f16(false, av, false, bv,
                                                     (short)0, acc, false, false);
    }
    #pragma unroll
    for (int r = 0; r < 8; r++) {
        const int m = r + ((lane >> 4) << 3);
        if (m < BB) logits[m][j0 + N] = acc[r] + fc_b[j0 + N];
    }

    // ---- GEMM2 (wave 0): z = relu(gap @ se_w1^T + se_b1), 16 hidden units
    if (wave == 0) {
        v8f acc2 = {};
        for (int kk = 0; kk < CC; kk += 32) {
            v16h av, bv;
            #pragma unroll
            for (int e = 0; e < 16; e++) {
                av[e] = Asm[M][kk + kbA + (e & 7) + ((e >> 3) << 4)];
                bv[e] = (_Float16)se_w1[N * CC + kk + kbB + (e & 7) + ((e >> 3) << 4)];
            }
            acc2 = __builtin_amdgcn_wmma_f32_16x16x32_f16(false, av, false, bv,
                                                          (short)0, acc2, false, false);
        }
        #pragma unroll
        for (int r = 0; r < 8; r++) {
            const int m = r + ((lane >> 4) << 3);
            if (m < BB) Zsm[m][N] = (_Float16)fmaxf(acc2[r] + se_b1[N], 0.f);
        }
    }
    __syncthreads();

    // ---- softmax over channels -> wsoft (threads 0..3, tiny)
    if (tid < BB) {
        float mx = -1e30f;
        for (int j = 0; j < CC; j++) mx = fmaxf(mx, logits[tid][j]);
        float s = 0.f;
        for (int j = 0; j < CC; j++) s += expf(logits[tid][j] - mx);
        const float inv = 1.f / s;
        for (int j = 0; j < CC; j++)
            wsoft[tid * CC + j] = expf(logits[tid][j] - mx) * inv;
    }

    // ---- GEMM3: w_se = sigmoid(z @ se_w2^T + se_b2); K=32 with zero pad
    {
        v8f acc3 = {};
        v16h av, bv;
        #pragma unroll
        for (int e = 0; e < 16; e++) {
            av[e] = Zsm[M][kbA + (e & 7) + ((e >> 3) << 4)];
            const int k = kbB + (e & 7) + ((e >> 3) << 4);
            bv[e] = (k < HIDDEN) ? (_Float16)se_w2[(j0 + N) * HIDDEN + k]
                                 : (_Float16)0.f;
        }
        acc3 = __builtin_amdgcn_wmma_f32_16x16x32_f16(false, av, false, bv,
                                                      (short)0, acc3, false, false);
        #pragma unroll
        for (int r = 0; r < 8; r++) {
            const int m = r + ((lane >> 4) << 3);
            if (m < BB) {
                const float v = acc3[r] + se_b2[j0 + N];
                wse[m * CC + j0 + N] = sigmoidf_(v);
            }
        }
    }
}

// ---------------------------------------------------------------------------
// Kernel E: fused final pass.
//   Recompute bands (x is L2-resident), threshold/gate, mix with attn,
//   redistribute with softmax(sub_redistribute[c]), inverse Haar, and blend:
//   out = x * wsoft[b,c] + gamma * X_re * wse[b,c]
// One block per plane; per-plane scalars staged once in LDS.
// ---------------------------------------------------------------------------
__global__ void hfp_kE_out(const float* __restrict__ x,
                           const float* __restrict__ theta,
                           const float* __restrict__ dgate,
                           const float* __restrict__ sr,
                           const float* __restrict__ msum,
                           const float* __restrict__ attn,
                           const float* __restrict__ wsoft,
                           const float* __restrict__ wse,
                           const float* __restrict__ gptr,
                           float* __restrict__ out) {
    const int p = blockIdx.x;
    const int b = p >> 7;
    const int c = p & 127;
    __shared__ float sc[12];
    if (threadIdx.x == 0) {
        #pragma unroll
        for (int k = 0; k < 3; k++) {
            sc[k]     = sigmoidf_(theta[k * CC + c]) *
                        (msum[k * BC + p] * (1.f / (float)NBLK) + 1e-6f);
            sc[3 + k] = sigmoidf_(dgate[k * CC + c]);
        }
        const float r0 = sr[c * 3 + 0], r1 = sr[c * 3 + 1], r2 = sr[c * 3 + 2];
        const float mx = fmaxf(r0, fmaxf(r1, r2));
        const float e0 = expf(r0 - mx), e1 = expf(r1 - mx), e2 = expf(r2 - mx);
        const float inv = 1.f / (e0 + e1 + e2);
        sc[6] = e0 * inv; sc[7] = e1 * inv; sc[8] = e2 * inv;
        sc[9] = wsoft[p]; sc[10] = wse[p]; sc[11] = gptr[0];
    }
    __syncthreads();
    const float thr0 = sc[0], thr1 = sc[1], thr2 = sc[2];
    const float g0 = sc[3], g1 = sc[4], g2 = sc[5];
    const float wr0 = sc[6], wr1 = sc[7], wr2 = sc[8];
    const float wsC = sc[9], wseC = sc[10], gamma = sc[11];
    // inverse-Haar mixing coefficients for the redistributed H_mix
    const float ka = -wr0 - wr1 + wr2;
    const float kb =  wr0 - wr1 - wr2;
    const float kc = -wr0 + wr1 - wr2;
    const float kd =  wr0 + wr1 + wr2;

    const float* xp = x + (size_t)p * PLANE;
    float* op = out + (size_t)p * PLANE;
    const float* at0 = attn + (size_t)(0 * BB + b) * NBLK;
    const float* at1 = attn + (size_t)(1 * BB + b) * NBLK;
    const float* at2 = attn + (size_t)(2 * BB + b) * NBLK;

    for (int i = threadIdx.x; i < NBLK; i += 256) {
        const int h2 = i >> 7, w2 = i & 127;
        const int off = (2 * h2) * WW + 2 * w2;
        const float2 t0 = *(const float2*)(xp + off);
        const float2 t1 = *(const float2*)(xp + off + WW);
        const float a = t0.x, bb_ = t0.y, cc_ = t1.x, dd = t1.y;
        const float LL  = 0.5f * ( a + bb_ + cc_ + dd);
        const float LH  = 0.5f * (-a + bb_ - cc_ + dd);
        const float HL  = 0.5f * (-a - bb_ + cc_ + dd);
        const float HHv = 0.5f * ( a - bb_ - cc_ + dd);
        const float h0 = copysignf(fmaxf(fabsf(LH)  - thr0, 0.f), LH)  * g0;
        const float h1 = copysignf(fmaxf(fabsf(HL)  - thr1, 0.f), HL)  * g1;
        const float h2v = copysignf(fmaxf(fabsf(HHv) - thr2, 0.f), HHv) * g2;
        const float Hmix = h0 * at0[i] + h1 * at1[i] + h2v * at2[i];
        const float ap = 0.5f * (LL + Hmix * ka);
        const float bp = 0.5f * (LL + Hmix * kb);
        const float cp = 0.5f * (LL + Hmix * kc);
        const float dp = 0.5f * (LL + Hmix * kd);
        float2 o0, o1;
        o0.x = a   * wsC + gamma * ap * wseC;
        o0.y = bb_ * wsC + gamma * bp * wseC;
        o1.x = cc_ * wsC + gamma * cp * wseC;
        o1.y = dd  * wsC + gamma * dp * wseC;
        *(float2*)(op + off)      = o0;
        *(float2*)(op + off + WW) = o1;
    }
}

// ---------------------------------------------------------------------------
// Launcher
// ---------------------------------------------------------------------------
extern "C" void kernel_launch(void* const* d_in, const int* in_sizes, int n_in,
                              void* d_out, int out_size, void* d_ws, size_t ws_size,
                              hipStream_t stream) {
    const float* x      = (const float*)d_in[0];
    const float* theta  = (const float*)d_in[1];
    const float* dgate  = (const float*)d_in[2];
    const float* fc_w   = (const float*)d_in[3];
    const float* fc_b   = (const float*)d_in[4];
    const float* sr     = (const float*)d_in[5];
    const float* se_w1  = (const float*)d_in[6];
    const float* se_b1  = (const float*)d_in[7];
    const float* se_w2  = (const float*)d_in[8];
    const float* se_b2  = (const float*)d_in[9];
    const float* gamma  = (const float*)d_in[10];
    float* out = (float*)d_out;

    float* ws    = (float*)d_ws;
    float* msum  = ws;           // 3*512  = 1536 floats
    float* gsum  = ws + 1536;    // 512
    float* wsoft = ws + 2048;    // 512
    float* wse   = ws + 2560;    // 512
    float* attn  = ws + 3072;    // 3*4*16384 = 196608 floats (~786 KB)

    hfp_kA_plane_stats<<<BC, 256, 0, stream>>>(x, msum, gsum);
    hfp_kB_attn<<<BB * 64, 256, 0, stream>>>(x, theta, dgate, msum, attn);
    hfp_kD_head<<<1, 256, 0, stream>>>(gsum, fc_w, fc_b, se_w1, se_b1,
                                       se_w2, se_b2, wsoft, wse);
    hfp_kE_out<<<BC, 256, 0, stream>>>(x, theta, dgate, sr, msum, attn,
                                       wsoft, wse, gamma, out);
}
